// Qwen3Attention_17660905521379
// MI455X (gfx1250) — matrix operand order repaired and checked by reading, not compiled, not run on hardware
//
#include <hip/hip_runtime.h>
#include <hip/hip_bf16.h>

// ---- problem constants (match reference) ----
#define T_SEQ   2048
#define D_MODEL 4096
#define NHQ     32
#define NKVH    8
#define HDIM    128
#define QSTRIDE (NHQ * HDIM)   // 4096
#define KSTRIDE (NKVH * HDIM)  // 1024
#define EPS_F   1e-6f
#define LOG_THETA 13.815510557964274f   // ln(1e6)
#define INV_SQRT_HD 0.08838834764831845f

typedef __attribute__((ext_vector_type(16))) __bf16 v16bf;
typedef __attribute__((ext_vector_type(8)))  float  v8f;
typedef __attribute__((ext_vector_type(4)))  int    v4i_t;

union FragBF { uint4 u4[2]; v16bf v; };

#if defined(__AMDGCN__) && __has_builtin(__builtin_amdgcn_global_load_async_to_lds_b128)
#define HAS_ASYNC_LDS 1
#else
#define HAS_ASYNC_LDS 0
#endif

__device__ __forceinline__ unsigned short f2bf(float f) {
  union { float f; unsigned u; } x; x.f = f;
  unsigned r = x.u + 0x7FFFu + ((x.u >> 16) & 1u);   // round-to-nearest-even
  return (unsigned short)(r >> 16);
}

__device__ __forceinline__ void lds_fence() {
  asm volatile("s_wait_dscnt 0" ::: "memory");       // CDNA5 split DS counter
}

// Async global->LDS 16B copy (ASYNCcnt path), fallback = VGPR round trip.
__device__ __forceinline__ void async_copy16(void* lds_dst, const void* gsrc) {
#if HAS_ASYNC_LDS
  __builtin_amdgcn_global_load_async_to_lds_b128(
      (__attribute__((address_space(1))) v4i_t*)gsrc,
      (__attribute__((address_space(3))) v4i_t*)lds_dst, 0, 0);
#else
  *(uint4*)lds_dst = *(const uint4*)gsrc;
#endif
}
__device__ __forceinline__ void async_fence() {
#if HAS_ASYNC_LDS
  asm volatile("s_wait_asynccnt 0" ::: "memory");
#endif
}

__device__ __forceinline__ v8f wmma_bf16(const FragBF& a, const FragBF& b, v8f c) {
  return __builtin_amdgcn_wmma_f32_16x16x32_bf16(false, a.v, false, b.v,
                                                 (short)0, c, false, false);
}

// ============================================================
// Blocked GEMM: C[M,N] f32 = A[M,K] * opB.  Block tile 128x128,
// 8 waves, each wave owns 32x64 (2x4 WMMA tiles, 8 accumulators).
// K staged in LDS 32 at a time, shared by all waves.
// TRANS_B=0: B is [K,N] f32.  TRANS_B=1: B is [N,K] f32.
// A_IS_BF16: A already bf16 (staged via async-to-LDS), else f32.
// ============================================================
template<bool TRANS_B, bool A_IS_BF16>
__global__ __launch_bounds__(256)
void gemm_wmma_kernel(const void* __restrict__ Av, const float* __restrict__ B,
                      float* __restrict__ C, int M, int N, int K,
                      int lda, int ldb, int ldc) {
  __shared__ unsigned short A_s[128][32];   // [row][k]   8 KiB
  __shared__ unsigned short B_s[32][128];   // [k][col]   8 KiB

  const int tid  = threadIdx.x;
  const int lane = tid & 31;
  const int w    = tid >> 5;
  const int nbBlocks = N >> 7;
  const int m0 = (blockIdx.x / nbBlocks) << 7;
  const int n0 = (blockIdx.x % nbBlocks) << 7;
  const int m  = lane & 15;
  const int hi = lane >> 4;
  const int wr = w >> 1;          // wave row group: rows wr*32 .. +31
  const int wc = w & 1;           // wave col group: cols wc*64 .. +63

  v8f acc[2][4];
  #pragma unroll
  for (int i = 0; i < 2; ++i)
    #pragma unroll
    for (int j = 0; j < 4; ++j)
      acc[i][j] = (v8f){0.f,0.f,0.f,0.f,0.f,0.f,0.f,0.f};

  for (int k0 = 0; k0 < K; k0 += 32) {
    // ---- stage A panel 128x32 ----
    {
      const int row  = tid >> 1;          // 0..127
      const int half = tid & 1;           // k halves of 16
      if (A_IS_BF16) {
        const unsigned short* A = (const unsigned short*)Av;
        const unsigned short* src = &A[(size_t)(m0 + row) * lda + k0 + half * 16];
        async_copy16(&A_s[row][half * 16],     src);
        async_copy16(&A_s[row][half * 16 + 8], src + 8);
        __builtin_prefetch(src + 64, 0, 0);
      } else {
        const float* A = (const float*)Av;
        const float4* src = (const float4*)&A[(size_t)(m0 + row) * lda + k0 + half * 16];
        unsigned short tmp[16];
        #pragma unroll
        for (int q = 0; q < 4; ++q) {
          float4 f = src[q];
          tmp[q*4+0] = f2bf(f.x); tmp[q*4+1] = f2bf(f.y);
          tmp[q*4+2] = f2bf(f.z); tmp[q*4+3] = f2bf(f.w);
        }
        uint4* dst = (uint4*)&A_s[row][half * 16];
        dst[0] = *(const uint4*)&tmp[0];
        dst[1] = *(const uint4*)&tmp[8];
        __builtin_prefetch(src + 8, 0, 0);   // next k-panel, same row
      }
    }
    // ---- stage B panel 32x128 ----
    if (!TRANS_B) {
      const int k   = tid >> 3;            // 0..31
      const int seg = (tid & 7) << 4;      // 0..112
      const float4* src = (const float4*)&B[(size_t)(k0 + k) * ldb + n0 + seg];
      unsigned short tmp[16];
      #pragma unroll
      for (int q = 0; q < 4; ++q) {
        float4 f = src[q];
        tmp[q*4+0] = f2bf(f.x); tmp[q*4+1] = f2bf(f.y);
        tmp[q*4+2] = f2bf(f.z); tmp[q*4+3] = f2bf(f.w);
      }
      uint4* dst = (uint4*)&B_s[k][seg];
      dst[0] = *(const uint4*)&tmp[0];
      dst[1] = *(const uint4*)&tmp[8];
      __builtin_prefetch((const float*)src + 32 * ldb, 0, 0);
    } else {
      const int n    = tid >> 1;           // output column 0..127
      const int half = tid & 1;
      const float4* src = (const float4*)&B[(size_t)(n0 + n) * ldb + k0 + half * 16];
      #pragma unroll
      for (int q = 0; q < 4; ++q) {
        float4 f = src[q];
        B_s[half*16 + q*4 + 0][n] = f2bf(f.x);
        B_s[half*16 + q*4 + 1][n] = f2bf(f.y);
        B_s[half*16 + q*4 + 2][n] = f2bf(f.z);
        B_s[half*16 + q*4 + 3][n] = f2bf(f.w);
      }
      __builtin_prefetch((const float*)src + 8, 0, 0);
    }
    async_fence();
    __syncthreads();

    // ---- consume: 8 WMMAs per wave per K-step ----
    FragBF fa[2];
    #pragma unroll
    for (int mt = 0; mt < 2; ++mt) {
      const int row = wr * 32 + mt * 16 + m;
      fa[mt].u4[0] = *(const uint4*)&A_s[row][hi * 8];
      fa[mt].u4[1] = *(const uint4*)&A_s[row][16 + hi * 8];
    }
    #pragma unroll
    for (int nt = 0; nt < 4; ++nt) {
      FragBF fb;
      const int cb = wc * 64 + nt * 16;
      #pragma unroll
      for (int s = 0; s < 16; ++s)
        ((unsigned short*)&fb)[s] = B_s[(s < 8 ? hi * 8 + s : 8 + hi * 8 + s)][cb + m];
      acc[0][nt] = wmma_bf16(fa[0], fb, acc[0][nt]);
      acc[1][nt] = wmma_bf16(fa[1], fb, acc[1][nt]);
    }
    __syncthreads();
  }

  #pragma unroll
  for (int mt = 0; mt < 2; ++mt)
    #pragma unroll
    for (int nt = 0; nt < 4; ++nt)
      #pragma unroll
      for (int r = 0; r < 8; ++r)
        C[(size_t)(m0 + wr * 32 + mt * 16 + r + hi * 8) * ldc +
          n0 + wc * 64 + nt * 16 + m] = acc[mt][nt][r];
}

// ============================================================
// RMSNorm (fp32) + neox RoPE, one wave per (t, head). Emits bf16.
// ============================================================
__global__ __launch_bounds__(256)
void norm_rope_kernel(const float* __restrict__ in, const float* __restrict__ scale,
                      const int* __restrict__ positions,
                      unsigned short* __restrict__ out, int heads) {
  const int lane = threadIdx.x & 31;
  const int wid  = (blockIdx.x << 3) + (threadIdx.x >> 5);
  const int t = wid / heads;
  const int h = wid % heads;
  const float* x = in + (size_t)(t * heads + h) * HDIM;

  float v0 = x[lane], v1 = x[lane + 32], v2 = x[lane + 64], v3 = x[lane + 96];
  float ss = v0*v0 + v1*v1 + v2*v2 + v3*v3;
  #pragma unroll
  for (int d = 16; d >= 1; d >>= 1) ss += __shfl_xor(ss, d, 32);
  float r = rsqrtf(ss * (1.0f / HDIM) + EPS_F);

  float n0 = v0 * r * scale[lane];
  float n1 = v1 * r * scale[lane + 32];
  float n2 = v2 * r * scale[lane + 64];
  float n3 = v3 * r * scale[lane + 96];

  float pos = (float)positions[t];
  float a0 = pos * __expf(-((float)lane)        * (1.0f / 64.0f) * LOG_THETA);
  float a1 = pos * __expf(-((float)(lane + 32)) * (1.0f / 64.0f) * LOG_THETA);
  float c0, s0, c1, s1;
  __sincosf(a0, &s0, &c0);
  __sincosf(a1, &s1, &c1);

  unsigned short* y = out + (size_t)(t * heads + h) * HDIM;
  y[lane]      = f2bf(n0 * c0 - n2 * s0);
  y[lane + 32] = f2bf(n1 * c1 - n3 * s1);
  y[lane + 64] = f2bf(n2 * c0 + n0 * s0);
  y[lane + 96] = f2bf(n3 * c1 + n1 * s1);
}

__global__ __launch_bounds__(256)
void cvt_f2bf_kernel(const float* __restrict__ in, unsigned short* __restrict__ out, int n) {
  int i = blockIdx.x * 256 + threadIdx.x;
  if (i < n) out[i] = f2bf(in[i]);
}

// ============================================================
// Flash attention (causal GQA): 8 waves/block share one head and
// 128 Q rows; K tile transposed into LDS, V tile async-copied.
// ============================================================
__global__ __launch_bounds__(256)
void attn_kernel(const unsigned short* __restrict__ qbf,
                 const unsigned short* __restrict__ kbf,
                 const unsigned short* __restrict__ vbf,
                 unsigned short* __restrict__ attn_out) {
  __shared__ unsigned short kT[HDIM][32];      // [h][key]
  __shared__ unsigned short v_s[32][HDIM];     // [key][h]
  __shared__ unsigned short p_s[8][16][32];    // per-wave P tile

  const int lane   = threadIdx.x & 31;
  const int w      = threadIdx.x >> 5;
  const int head   = blockIdx.x >> 4;
  const int tbase  = (blockIdx.x & 15) << 7;   // 128 Q rows per block
  const int t0     = tbase + (w << 4);
  const int kv     = head >> 2;                // G = 4
  const int m      = lane & 15;
  const int hi     = lane >> 4;

  // Q fragments: 4 k-chunks of 32 along HD
  FragBF aq[4];
  {
    const unsigned short* qrow = qbf + (size_t)(t0 + m) * QSTRIDE + head * HDIM;
    #pragma unroll
    for (int c = 0; c < 4; ++c) {
      aq[c].u4[0] = *(const uint4*)&qrow[c * 32 + hi * 8];
      aq[c].u4[1] = *(const uint4*)&qrow[c * 32 + 16 + hi * 8];
    }
  }

  float Mx[8], L[8];
  v8f O[8];
  #pragma unroll
  for (int r = 0; r < 8; ++r) { Mx[r] = -1e30f; L[r] = 0.f; }
  #pragma unroll
  for (int j = 0; j < 8; ++j) O[j] = (v8f){0.f,0.f,0.f,0.f,0.f,0.f,0.f,0.f};

  const int smax = tbase + 128;
  for (int s0 = 0; s0 < smax; s0 += 32) {
    __syncthreads();
    { // stage V (async memory->LDS) and K (transposed via VGPRs)
      const int row = threadIdx.x >> 3;
      const int seg = (threadIdx.x & 7) << 4;
      const unsigned short* vp = vbf + (size_t)(s0 + row) * KSTRIDE + kv * HDIM + seg;
      async_copy16(&v_s[row][seg],     vp);
      async_copy16(&v_s[row][seg + 8], vp + 8);
      const unsigned short* kp = kbf + (size_t)(s0 + row) * KSTRIDE + kv * HDIM + seg;
      unsigned short tmp[16];
      *(uint4*)&tmp[0] = *(const uint4*)&kp[0];
      *(uint4*)&tmp[8] = *(const uint4*)&kp[8];
      #pragma unroll
      for (int j = 0; j < 16; ++j) kT[seg + j][row] = tmp[j];
      __builtin_prefetch(kp + 32 * KSTRIDE, 0, 0);
      __builtin_prefetch(vp + 32 * KSTRIDE, 0, 0);
    }
    async_fence();
    __syncthreads();

    if (s0 > t0 + 15) continue;   // wave-uniform: keeps EXEC all-1s for WMMA

    // S = Q * K^T  (two 16-key column tiles)
    v8f S0 = (v8f){0.f,0.f,0.f,0.f,0.f,0.f,0.f,0.f};
    v8f S1 = S0;
    #pragma unroll
    for (int c = 0; c < 4; ++c) {
      FragBF bk0, bk1;
      #pragma unroll
      for (int s = 0; s < 16; ++s) {
        const int kk = c * 32 + (s < 8 ? hi * 8 + s : 8 + hi * 8 + s);
        ((unsigned short*)&bk0)[s] = kT[kk][m];
        ((unsigned short*)&bk1)[s] = kT[kk][16 + m];
      }
      S0 = wmma_bf16(aq[c], bk0, S0);
      S1 = wmma_bf16(aq[c], bk1, S1);
    }

    // online softmax per row (rows replicated across 16-lane halves)
    float alpha[8];
    #pragma unroll
    for (int r = 0; r < 8; ++r) {
      const int trow = t0 + r + hi * 8;
      float x0 = (s0 + m      <= trow) ? S0[r] * INV_SQRT_HD : -1e30f;
      float x1 = (s0 + 16 + m <= trow) ? S1[r] * INV_SQRT_HD : -1e30f;
      float mx = fmaxf(x0, x1);
      #pragma unroll
      for (int d = 8; d >= 1; d >>= 1) mx = fmaxf(mx, __shfl_xor(mx, d, 32));
      float Mn = fmaxf(Mx[r], mx);
      float a  = __expf(Mx[r] - Mn);
      float p0 = __expf(x0 - Mn);
      float p1 = __expf(x1 - Mn);
      float rs = p0 + p1;
      #pragma unroll
      for (int d = 8; d >= 1; d >>= 1) rs += __shfl_xor(rs, d, 32);
      L[r] = L[r] * a + rs;
      Mx[r] = Mn;
      alpha[r] = a;
      p_s[w][r + hi * 8][m]      = f2bf(p0);
      p_s[w][r + hi * 8][16 + m] = f2bf(p1);
    }
    lds_fence();

    // P back as A-fragment, then O += P * V
    FragBF ap;
    ap.u4[0] = *(const uint4*)&p_s[w][m][hi * 8];
    ap.u4[1] = *(const uint4*)&p_s[w][m][16 + hi * 8];
    #pragma unroll
    for (int j = 0; j < 8; ++j) {
      #pragma unroll
      for (int r = 0; r < 8; ++r) O[j][r] *= alpha[r];
      FragBF bv;
      #pragma unroll
      for (int s = 0; s < 16; ++s)
        ((unsigned short*)&bv)[s] = v_s[(s < 8 ? hi * 8 + s : 8 + hi * 8 + s)][j * 16 + m];
      O[j] = wmma_bf16(ap, bv, O[j]);
    }
    lds_fence();
  }

  #pragma unroll
  for (int j = 0; j < 8; ++j)
    #pragma unroll
    for (int r = 0; r < 8; ++r)
      attn_out[(size_t)(t0 + r + hi * 8) * QSTRIDE + head * HDIM + j * 16 + m] =
          f2bf(O[j][r] / L[r]);
}

// ============================================================
extern "C" void kernel_launch(void* const* d_in, const int* in_sizes, int n_in,
                              void* d_out, int out_size, void* d_ws, size_t ws_size,
                              hipStream_t stream) {
  (void)in_sizes; (void)n_in; (void)out_size; (void)ws_size;
  const float* x        = (const float*)d_in[0];
  const int*   positions= (const int*)  d_in[1];
  const float* Wq       = (const float*)d_in[2];   // [4096 out, 4096 in] rows = out cols
  const float* q_scale  = (const float*)d_in[3];
  const float* Wk       = (const float*)d_in[4];   // [4096, 1024]
  const float* k_scale  = (const float*)d_in[5];
  const float* Wv       = (const float*)d_in[6];   // [4096, 1024]
  const float* Wo       = (const float*)d_in[7];   // [4096, 4096]
  float* out = (float*)d_out;

  char* ws = (char*)d_ws;
  float*          q_f  = (float*)(ws + 0);                         // 32 MiB
  float*          k_f  = (float*)(ws + (size_t)33554432);          //  8 MiB
  float*          v_f  = (float*)(ws + (size_t)41943040);          //  8 MiB
  unsigned short* q_bf = (unsigned short*)(ws + (size_t)50331648); // 16 MiB
  unsigned short* k_bf = (unsigned short*)(ws + (size_t)67108864); //  4 MiB
  unsigned short* v_bf = (unsigned short*)(ws + (size_t)71303168); //  4 MiB
  unsigned short* a_bf = (unsigned short*)(ws + (size_t)75497472); // 16 MiB

  // Q = x @ Wq^T : 16x32 blocks of 128x128
  gemm_wmma_kernel<true , false><<<512, 256, 0, stream>>>(x, Wq, q_f,
      T_SEQ, QSTRIDE, D_MODEL, D_MODEL, D_MODEL, QSTRIDE);
  // K = x @ Wk ; V = x @ Wv : 16x8 blocks
  gemm_wmma_kernel<false, false><<<128, 256, 0, stream>>>(x, Wk, k_f,
      T_SEQ, KSTRIDE, D_MODEL, D_MODEL, KSTRIDE, KSTRIDE);
  gemm_wmma_kernel<false, false><<<128, 256, 0, stream>>>(x, Wv, v_f,
      T_SEQ, KSTRIDE, D_MODEL, D_MODEL, KSTRIDE, KSTRIDE);

  // RMSNorm + RoPE -> bf16
  norm_rope_kernel<<<8192, 256, 0, stream>>>(q_f, q_scale, positions, q_bf, NHQ);
  norm_rope_kernel<<<2048, 256, 0, stream>>>(k_f, k_scale, positions, k_bf, NKVH);
  cvt_f2bf_kernel<<<8192, 256, 0, stream>>>(v_f, v_bf, T_SEQ * KSTRIDE);

  // causal GQA flash attention -> bf16 attn
  attn_kernel<<<NHQ * (T_SEQ / 128), 256, 0, stream>>>(q_bf, k_bf, v_bf, a_bf);

  // O = attn @ Wo -> d_out (fp32), A is bf16 (async-to-LDS staging)
  gemm_wmma_kernel<false, true><<<512, 256, 0, stream>>>(a_bf, Wo, out,
      T_SEQ, D_MODEL, QSTRIDE, QSTRIDE, D_MODEL, D_MODEL);
}
